// CapsuleNet_33320356283013
// MI455X (gfx1250) — matrix operand order repaired and checked by reading, not compiled, not run on hardware
//
#include <hip/hip_runtime.h>
#include <hip/hip_bf16.h>

// ---------------------------------------------------------------------------
// CapsNet forward for MI455X (gfx1250, wave32, WMMA).
// Dominant op: primary-caps conv as implicit GEMM (M=18432,N=256,K=20736),
// bf16 hi/lo split -> 3x v_wmma_f32_16x16x32_bf16 per K-step (~fp32 accuracy,
// f32 accumulate in the WMMA). Double-buffered LDS tiles staged with
// GLOBAL_LOAD_ASYNC_TO_LDS_B128 (ASYNCcnt) so global fetch overlaps WMMA.
// ---------------------------------------------------------------------------

typedef __attribute__((ext_vector_type(16))) __bf16 v16bf;
typedef __attribute__((ext_vector_type(8)))  __bf16 bf16x8;
typedef __attribute__((ext_vector_type(8)))  float   v8f;
typedef __attribute__((ext_vector_type(4)))  unsigned int u32x4;
typedef __attribute__((ext_vector_type(4)))  int         v4i;

#define LDK 40   // padded LDS row stride (elements): 80B, 16B-aligned, conflict-free

#define AS1 __attribute__((address_space(1)))
#define AS3 __attribute__((address_space(3)))

#if __has_builtin(__builtin_amdgcn_global_load_async_to_lds_b128) && \
    __has_builtin(__builtin_amdgcn_s_wait_asynccnt)
#define HAVE_ASYNC 1
#else
#define HAVE_ASYNC 0
#endif

// one 16-byte lane-transfer: async DMA to LDS if available, else sync copy
__device__ __forceinline__ void cp16(const unsigned short* g, unsigned short* l) {
#if HAVE_ASYNC
    __builtin_amdgcn_global_load_async_to_lds_b128(
        (AS1 v4i*)(unsigned long long)(const void*)g,   // global src (int4*, AS1)
        (AS3 v4i*)(void*)l,                             // LDS dst   (int4*, AS3)
        0, 0);
#else
    *(u32x4*)l = *(const u32x4*)g;
#endif
}

// round-to-nearest-even fp32 -> bf16 bits
__device__ __forceinline__ unsigned short f2bfbits(float f) {
    unsigned u = __float_as_uint(f);
    unsigned r = u + 0x7fffu + ((u >> 16) & 1u);
    return (unsigned short)(r >> 16);
}

// ---------------------------------------------------------------------------
// Kernel 1: conv stem 28x28 -> 20x20x256, ReLU; write NHWC bf16 hi/lo planes.
// grid = 512*20 (one block per (batch, out-row)), 256 threads (= out channel).
// ---------------------------------------------------------------------------
__global__ __launch_bounds__(256) void k_conv0(
    const float* __restrict__ x, const float* __restrict__ w,
    const float* __restrict__ bias,
    unsigned short* __restrict__ ahi, unsigned short* __restrict__ alo)
{
    __shared__ float xs[9 * 28];
    const int b = blockIdx.x / 20;
    const int y = blockIdx.x % 20;
    const int o = threadIdx.x;

    for (int idx = threadIdx.x; idx < 9 * 28; idx += 256) {
        int ky = idx / 28, ix = idx % 28;
        xs[idx] = x[b * 784 + (y + ky) * 28 + ix];
    }
    float wreg[81];
#pragma unroll
    for (int p = 0; p < 81; ++p) wreg[p] = w[o * 81 + p];
    const float bv = bias[o];
    __syncthreads();

    for (int ox = 0; ox < 20; ++ox) {
        float acc = bv;
#pragma unroll
        for (int ky = 0; ky < 9; ++ky)
#pragma unroll
            for (int kx = 0; kx < 9; ++kx)
                acc = fmaf(xs[ky * 28 + ox + kx], wreg[ky * 9 + kx], acc);
        acc = fmaxf(acc, 0.f);
        unsigned short hb = f2bfbits(acc);
        float hf = __uint_as_float(((unsigned)hb) << 16);
        unsigned short lb = f2bfbits(acc - hf);
        int addr = (b * 400 + y * 20 + ox) * 256 + o;   // NHWC, C contiguous
        ahi[addr] = hb;
        alo[addr] = lb;
    }
}

// ---------------------------------------------------------------------------
// Kernel 2: repack prim_w (o,i,9,9) fp32 -> [pos][o][i] bf16 hi/lo planes.
// ---------------------------------------------------------------------------
__global__ __launch_bounds__(256) void k_wprep(
    const float* __restrict__ w,
    unsigned short* __restrict__ whi, unsigned short* __restrict__ wlo)
{
    int t = blockIdx.x * 256 + threadIdx.x;          // t = pos*65536 + o*256 + i
    int i = t & 255;
    int o = (t >> 8) & 255;
    int pos = t >> 16;
    float v = w[(o * 256 + i) * 81 + pos];
    unsigned short hb = f2bfbits(v);
    float hf = __uint_as_float(((unsigned)hb) << 16);
    unsigned short lb = f2bfbits(v - hf);
    whi[t] = hb;                                      // dst index == t
    wlo[t] = lb;
}

// ---------------------------------------------------------------------------
// WMMA fragment loads from LDS (row stride LDK=40 elements).
// A frag (16x32 bf16): lane half 0 -> K 0-7 & 16-23, half 1 -> K 8-15 & 24-31.
// B frag (32x16 bf16): lane half 0 -> K 0-7 & 16-23, half 1 -> K 8-15 & 24-31.
// ---------------------------------------------------------------------------
__device__ __forceinline__ v16bf ldfragA(const unsigned short* s, int row, int koff) {
    const bf16x8* p = (const bf16x8*)(s + row * LDK + koff);
    bf16x8 c0 = p[0];
    bf16x8 c1 = p[2];   // +16 elements
    v16bf r;
#pragma unroll
    for (int i = 0; i < 8; ++i) { r[i] = c0[i]; r[8 + i] = c1[i]; }
    return r;
}
__device__ __forceinline__ v16bf ldfragB(const unsigned short* s, int row, int koff) {
    const bf16x8* p = (const bf16x8*)(s + row * LDK + koff);
    bf16x8 c0 = p[0];
    bf16x8 c1 = p[2];   // +16 elements
    v16bf r;
#pragma unroll
    for (int i = 0; i < 8; ++i) { r[i] = c0[i]; r[8 + i] = c1[i]; }
    return r;
}

// ---------------------------------------------------------------------------
// Kernel 3: implicit-GEMM primary-caps conv via WMMA, double-buffered async
// staging. grid = 288 blocks (M tiles of 64), 256 threads = 8 waves.
// Block tile 64x256, wave tile 32x64 -> acc[2][4] of 16x16 f32.
// K loop: 648 steps (81 taps x 8 chunks of 32 ch); 3 WMMA per (mt,nt).
// Dynamic LDS: 2 x (A hi/lo 64x40 + B hi/lo 256x40) bf16 = 102,400 B.
// LDS element offsets: Ah = buf*2560, Al = 5120 + buf*2560,
//                      Bh = 10240 + buf*10240, Bl = 30720 + buf*10240.
// ---------------------------------------------------------------------------
__global__ __launch_bounds__(256) void k_gemm(
    const unsigned short* __restrict__ ahi, const unsigned short* __restrict__ alo,
    const unsigned short* __restrict__ whi, const unsigned short* __restrict__ wlo,
    const float* __restrict__ pbias, float* __restrict__ uout)
{
    extern __shared__ unsigned short smem[];

    const int tid = threadIdx.x;
    const int blk = blockIdx.x;

    // A staging role: one (row, 8-elem chunk) per thread
    const int arow = tid >> 2;
    const int ach  = (tid & 3) << 3;
    const int m0   = blk * 64 + arow;
    const int ab   = m0 / 36;
    const int ar   = m0 - ab * 36;
    const int aoy  = ar / 6;
    const int aox  = ar - aoy * 6;
    const int actBase = ab * 400 * 256;
    const int iy0 = 2 * aoy, ix0 = 2 * aox;

    const int wave = tid >> 5, lane = tid & 31;
    const int wm = wave >> 2;          // 0..1 : 32-row group
    const int wn = wave & 3;           // 0..3 : 64-col group
    const int l16 = lane & 15;
    const int koffA = (lane < 16) ? 0 : 8;
    const int koffB = (lane < 16) ? 0 : 8;

    v8f acc[2][4];
#pragma unroll
    for (int mt = 0; mt < 2; ++mt)
#pragma unroll
        for (int nt = 0; nt < 4; ++nt)
#pragma unroll
            for (int e = 0; e < 8; ++e) acc[mt][nt][e] = 0.f;

    // stage K-step ks (ks = pos*8 + chunk) into LDS buffer buf:
    // 10 x 16B lane-transfers per thread (2 A planes + 8 B quarters)
    auto stage = [&](int ks, int buf) {
        unsigned short* Ahp = smem + buf * 2560;
        unsigned short* Alp = smem + 5120 + buf * 2560;
        unsigned short* Bhp = smem + 10240 + buf * 10240;
        unsigned short* Blp = smem + 30720 + buf * 10240;
        const int p  = ks >> 3;
        const int kc = (ks & 7) << 5;
        const int ky = p / 9;
        const int kx = p - ky * 9;
        const int aoff = actBase + ((iy0 + ky) * 20 + (ix0 + kx)) * 256 + ach + kc;
        cp16(ahi + aoff, Ahp + arow * LDK + ach);
        cp16(alo + aoff, Alp + arow * LDK + ach);
        const int wbase = (p << 16) + kc;   // pos*256*256 + kc
#pragma unroll
        for (int q = 0; q < 4; ++q) {
            int idx = tid + (q << 8);
            int br  = idx >> 2;
            int bc  = (idx & 3) << 3;
            int src = wbase + br * 256 + bc;
            cp16(whi + src, Bhp + br * LDK + bc);
            cp16(wlo + src, Blp + br * LDK + bc);
        }
    };

    stage(0, 0);                       // prologue: fill buffer 0
    for (int ks = 0; ks < 648; ++ks) {
        const int cur = ks & 1;
        if (ks + 1 < 648) {
            stage(ks + 1, cur ^ 1);    // prefetch next tile into other buffer
#if HAVE_ASYNC
            __builtin_amdgcn_s_wait_asynccnt(10);  // batch for `cur` landed
#endif
        } else {
#if HAVE_ASYNC
            __builtin_amdgcn_s_wait_asynccnt(0);
#endif
        }
        __syncthreads();               // publish buffer `cur` to all waves

        const unsigned short* Ahp = smem + cur * 2560;
        const unsigned short* Alp = smem + 5120 + cur * 2560;
        const unsigned short* Bhp = smem + 10240 + cur * 10240;
        const unsigned short* Blp = smem + 30720 + cur * 10240;

        v16bf a_h[2], a_l[2];
#pragma unroll
        for (int mt = 0; mt < 2; ++mt) {
            int rr = wm * 32 + mt * 16 + l16;
            a_h[mt] = ldfragA(Ahp, rr, koffA);
            a_l[mt] = ldfragA(Alp, rr, koffA);
        }
#pragma unroll
        for (int nt = 0; nt < 4; ++nt) {
            int nr = wn * 64 + nt * 16 + l16;
            v16bf b_h = ldfragB(Bhp, nr, koffB);
            v16bf b_l = ldfragB(Blp, nr, koffB);
#pragma unroll
            for (int mt = 0; mt < 2; ++mt) {
                acc[mt][nt] = __builtin_amdgcn_wmma_f32_16x16x32_bf16(
                    false, a_h[mt], false, b_h, (short)0, acc[mt][nt], false, false);
                acc[mt][nt] = __builtin_amdgcn_wmma_f32_16x16x32_bf16(
                    false, a_h[mt], false, b_l, (short)0, acc[mt][nt], false, false);
                acc[mt][nt] = __builtin_amdgcn_wmma_f32_16x16x32_bf16(
                    false, a_l[mt], false, b_h, (short)0, acc[mt][nt], false, false);
            }
        }
        __syncthreads();               // all waves done reading `cur` before reuse
    }

    // Epilogue: +bias, scatter to NCHW flat order -> matches reshape (B,1152,8)
    const int rowHalf = (lane >= 16) ? 8 : 0;
#pragma unroll
    for (int nt = 0; nt < 4; ++nt) {
        const int n_g = wn * 64 + nt * 16 + l16;
        const float bvv = pbias[n_g];
#pragma unroll
        for (int mt = 0; mt < 2; ++mt) {
#pragma unroll
            for (int j = 0; j < 8; ++j) {
                int m_g = blk * 64 + wm * 32 + mt * 16 + j + rowHalf;
                int bb = m_g / 36;
                int rr = m_g - bb * 36;
                uout[(bb * 256 + n_g) * 36 + rr] = acc[mt][nt][j] + bvv;
            }
        }
    }
}

// ---------------------------------------------------------------------------
// Kernel 4: squash (B,1152,8) in place.
// ---------------------------------------------------------------------------
__global__ __launch_bounds__(256) void k_squash(float* __restrict__ u)
{
    int c = blockIdx.x * 256 + threadIdx.x;   // < 512*1152
    float* p = u + c * 8;
    float v[8], s = 0.f;
#pragma unroll
    for (int i = 0; i < 8; ++i) { v[i] = p[i]; s = fmaf(v[i], v[i], s); }
    float n = sqrtf(s);
    float sc = s / ((1.f + s) * (n + 1e-8f));
#pragma unroll
    for (int i = 0; i < 8; ++i) p[i] = v[i] * sc;
}

// ---------------------------------------------------------------------------
// Kernel 5: x_hat[b,o,i] = dot8(digit_w[o,i,0,:], u[b,i,:])
// ---------------------------------------------------------------------------
__global__ __launch_bounds__(256) void k_xhat(
    const float* __restrict__ u, const float* __restrict__ dw,
    float* __restrict__ xh)
{
    int t = blockIdx.x * 256 + threadIdx.x;   // < 512*10*1152
    int i = t % 1152;
    int ob = t / 1152;
    int o = ob % 10;
    int b = ob / 10;
    const float* up = u + (b * 1152 + i) * 8;
    const float* wp = dw + (o * 1152 + i) * 8;
    float acc = 0.f;
#pragma unroll
    for (int k = 0; k < 8; ++k) acc = fmaf(up[k], wp[k], acc);
    xh[t] = acc;   // layout (b*10+o)*1152 + i
}

// ---------------------------------------------------------------------------
// Kernel 6: 3-iteration dynamic routing + output norm. One block per image.
// b_logits in LDS; each thread owns fixed i-columns -> only s needs barriers.
// ---------------------------------------------------------------------------
__global__ __launch_bounds__(256) void k_route(
    const float* __restrict__ xh, float* __restrict__ out)
{
    __shared__ float bl[10 * 1152];
    __shared__ float red[10 * 256];
    const int b = blockIdx.x, tid = threadIdx.x;

    float xc[5][10];
#pragma unroll
    for (int j = 0; j < 5; ++j) {
        int i = tid + j * 256;
        if (i < 1152)
            for (int o = 0; o < 10; ++o)
                xc[j][o] = xh[(b * 10 + o) * 1152 + i];
    }
    for (int idx = tid; idx < 10 * 1152; idx += 256) bl[idx] = 0.f;
    __syncthreads();

    float vout[10];
    for (int r = 0; r < 3; ++r) {
        float sp[10];
#pragma unroll
        for (int o = 0; o < 10; ++o) sp[o] = 0.f;
        for (int j = 0; j < 5; ++j) {
            int i = tid + j * 256;
            if (i < 1152) {
                float mx = -1e30f;
                for (int o = 0; o < 10; ++o) mx = fmaxf(mx, bl[o * 1152 + i]);
                float e[10], sum = 0.f;
                for (int o = 0; o < 10; ++o) { e[o] = expf(bl[o * 1152 + i] - mx); sum += e[o]; }
                float inv = 1.f / sum;
                for (int o = 0; o < 10; ++o) sp[o] += e[o] * inv * xc[j][o];
            }
        }
        __syncthreads();                       // previous red reads done
        for (int o = 0; o < 10; ++o) red[o * 256 + tid] = sp[o];
        __syncthreads();
        for (int off = 128; off; off >>= 1) {
            if (tid < off)
                for (int o = 0; o < 10; ++o)
                    red[o * 256 + tid] += red[o * 256 + tid + off];
            __syncthreads();
        }
        for (int o = 0; o < 10; ++o) {
            float s = red[o * 256];
            float n2 = s * s;
            float sc = n2 / ((1.f + n2) * (fabsf(s) + 1e-8f));
            vout[o] = sc * s;                  // squash of scalar
        }
        if (r < 2) {
            for (int j = 0; j < 5; ++j) {
                int i = tid + j * 256;
                if (i < 1152)
                    for (int o = 0; o < 10; ++o)
                        bl[o * 1152 + i] += vout[o] * xc[j][o];
            }
        }
    }
    if (tid < 10) out[b * 10 + tid] = fabsf(vout[tid]);
}

// ---------------------------------------------------------------------------
extern "C" void kernel_launch(void* const* d_in, const int* in_sizes, int n_in,
                              void* d_out, int out_size, void* d_ws, size_t ws_size,
                              hipStream_t stream)
{
    (void)in_sizes; (void)n_in; (void)out_size; (void)ws_size;
    const float* x   = (const float*)d_in[0];
    const float* c0w = (const float*)d_in[1];
    const float* c0b = (const float*)d_in[2];
    const float* pw  = (const float*)d_in[3];
    const float* pb  = (const float*)d_in[4];
    const float* dw  = (const float*)d_in[5];
    float* out = (float*)d_out;

    char* ws = (char*)d_ws;
    // workspace layout (bytes, all 256-aligned): total ~273.4 MB
    unsigned short* ahi = (unsigned short*)(ws);                 // 104,857,600
    unsigned short* alo = (unsigned short*)(ws + 104857600u);    // 104,857,600
    unsigned short* whi = (unsigned short*)(ws + 209715200u);    //  10,616,832
    unsigned short* wlo = (unsigned short*)(ws + 220332032u);    //  10,616,832
    float*          u   = (float*)(ws + 230948864u);             //  18,874,368
    float*          xh  = (float*)(ws + 249823232u);             //  23,592,960

    k_conv0 <<<512 * 20, 256, 0, stream>>>(x, c0w, c0b, ahi, alo);
    k_wprep <<<20736,     256, 0, stream>>>(pw, whi, wlo);
    k_gemm  <<<288,       256, 102400, stream>>>(ahi, alo, whi, wlo, pb, u);
    k_squash<<<2304,      256, 0, stream>>>(u);
    k_xhat  <<<23040,     256, 0, stream>>>(u, dw, xh);
    k_route <<<512,       256, 0, stream>>>(xh, out);
}
